// GNNPredictor_55611236549166
// MI455X (gfx1250) — matrix operand order repaired and checked by reading, not compiled, not run on hardware
//
#include <hip/hip_runtime.h>
#include <hip/hip_bf16.h>

typedef __attribute__((ext_vector_type(16))) __bf16 v16bf;
typedef __attribute__((ext_vector_type(8)))  float  v8f;

#define DEV __device__ __forceinline__
#define CDIV(a, b) (((a) + (b) - 1) / (b))
#define MSUB 5   // M sub-tiles per wave (80 rows); N/16=1250 divisible by 5
#define NSUB 2   // N sub-tiles per wave (32 cols); tilesN = 80/40/20 all even

DEV unsigned short f2bf(float f) {
  unsigned int u = __float_as_uint(f);
  unsigned int r = ((u >> 16) & 1u) + 0x7FFFu;   // round-to-nearest-even
  return (unsigned short)((u + r) >> 16);
}

union BV32 {            // 8 VGPRs worth of bf16 data (32 bytes)
  v16bf v;
  unsigned short s[16];
  uint4 q[2];
};

// ---------------- utility kernels ----------------
__global__ void k_fill(float* __restrict__ p, long n, float v) {
  long i = (long)blockIdx.x * blockDim.x + threadIdx.x;
  if (i < n) p[i] = v;
}

__global__ void k_deg(const int* __restrict__ dst, float* __restrict__ deg, int E) {
  int e = blockIdx.x * blockDim.x + threadIdx.x;
  if (e < E) atomicAdd(&deg[dst[e]], 1.0f);
}

__global__ void k_invsqrt(float* __restrict__ p, int n) {
  int i = blockIdx.x * blockDim.x + threadIdx.x;
  if (i < n) p[i] = 1.0f / sqrtf(p[i]);      // deg >= 1 (self loops)
}

__global__ void k_f2bf(const float* __restrict__ in, unsigned short* __restrict__ out, long n) {
  long i = (long)blockIdx.x * blockDim.x + threadIdx.x;
  if (i < n) out[i] = f2bf(in[i]);
}

// Pack row-major W[Din,Dout] (f32) into WMMA-B layout, tile-major [nb][kb] so that
// for a fixed output column tile, B advances by a constant 1024B per K-step:
// WP[((nb*tilesK+kb)*32 + lane)*16 + j] = bf16(W[kb*32 + (lane>>4)*8 + (j&7) + (j>>3)*16][nb*16 + (lane&15)])
__global__ void k_packW(const float* __restrict__ W, unsigned short* __restrict__ WP,
                        int Din, int Dout) {
  long idx = (long)blockIdx.x * blockDim.x + threadIdx.x;
  long total = (long)Din * Dout;
  if (idx >= total) return;
  int tile = (int)(idx >> 9);
  int r    = (int)(idx & 511);
  int lane = r >> 4;
  int j    = r & 15;
  int tilesK = Din >> 5;
  int nb = tile / tilesK;
  int kb = tile - nb * tilesK;
  int k   = (kb << 5) + ((lane >> 4) << 3) + (j & 7) + ((j >> 3) << 4);
  int col = (nb << 4) + (lane & 15);
  WP[idx] = f2bf(W[(size_t)k * Dout + col]);
}

// ---------------- WMMA GEMM, 80x32 output per wave ----------------
__global__ __launch_bounds__(256)
void k_gemm_bf16(const unsigned short* __restrict__ A,
                 const unsigned short* __restrict__ BP,
                 float* __restrict__ C, int Nrows, int K, int Nout) {
  const int lane   = threadIdx.x & 31;
  const int wid    = blockIdx.x * (blockDim.x >> 5) + (threadIdx.x >> 5);
  const int nPairs = Nout >> 5;                   // tilesN / NSUB
  const int groupsM = Nrows / (16 * MSUB);
  const int mG = wid / nPairs;
  const int nP = wid - mG * nPairs;
  if (mG >= groupsM) return;                      // wave-uniform: EXEC all-ones
  const int lm   = lane & 15;
  const int half = lane >> 4;
  const int nk   = K >> 5;

  const unsigned short* ar[MSUB];
#pragma unroll
  for (int s = 0; s < MSUB; ++s)
    ar[s] = A + (size_t)(mG * (16 * MSUB) + s * 16 + lm) * (size_t)K + (half << 3);
  const unsigned short* bq[NSUB];
#pragma unroll
  for (int t = 0; t < NSUB; ++t)
    bq[t] = BP + (((size_t)(nP * NSUB + t) * nk * 32 + lane) << 4);

  v8f acc[MSUB][NSUB];
#pragma unroll
  for (int s = 0; s < MSUB; ++s)
#pragma unroll
    for (int t = 0; t < NSUB; ++t) { v8f z = {}; acc[s][t] = z; }

  for (int kb = 0; kb < nk; ++kb) {
    // ---- load phase: put all B and A fragments in flight first ----
    BV32 bvv[NSUB];
#pragma unroll
    for (int t = 0; t < NSUB; ++t) {
      const uint4* bp = (const uint4*)(bq[t] + ((size_t)kb << 9));  // +1024B/step
      bvv[t].q[0] = bp[0];
      bvv[t].q[1] = bp[1];
    }
    BV32 avv[MSUB];
#pragma unroll
    for (int s = 0; s < MSUB; ++s) {
      avv[s].q[0] = *(const uint4*)(ar[s] + (kb << 5));        // K = kb*32+half*8..
      avv[s].q[1] = *(const uint4*)(ar[s] + (kb << 5) + 16);   // .. and +16
    }
    // ---- compute phase: 10 independent WMMAs ----
#pragma unroll
    for (int s = 0; s < MSUB; ++s)
#pragma unroll
      for (int t = 0; t < NSUB; ++t)
        acc[s][t] = __builtin_amdgcn_wmma_f32_16x16x32_bf16(
            false, avv[s].v, false, bvv[t].v, (short)0, acc[s][t], false, false);
  }

#pragma unroll
  for (int s = 0; s < MSUB; ++s) {
    const int mBase = mG * (16 * MSUB) + s * 16 + (half << 3);
#pragma unroll
    for (int t = 0; t < NSUB; ++t) {
      const int n = ((nP * NSUB + t) << 4) + lm;
#pragma unroll
      for (int v = 0; v < 8; ++v)
        C[(size_t)(mBase + v) * (size_t)Nout + n] = acc[s][t][v];
    }
  }
}

// fallback: single 16x16 tile per wave (remainder rows, or odd tilesN)
__global__ __launch_bounds__(256)
void k_gemm_bf16_1(const unsigned short* __restrict__ A,
                   const unsigned short* __restrict__ BP,
                   float* __restrict__ C, int Mrows, int K, int Nout) {
  const int lane   = threadIdx.x & 31;
  const int wid    = blockIdx.x * (blockDim.x >> 5) + (threadIdx.x >> 5);
  const int tilesN = Nout >> 4;
  const int tilesM = Mrows >> 4;
  const int mT = wid / tilesN;
  const int nT = wid - mT * tilesN;
  if (mT >= tilesM) return;
  const int lm   = lane & 15;
  const int half = lane >> 4;
  const int nk   = K >> 5;
  const unsigned short* arow = A + (size_t)(mT * 16 + lm) * (size_t)K + (half << 3);
  const unsigned short* bq = BP + (((size_t)nT * nk * 32 + lane) << 4);
  v8f acc = {};
  for (int kb = 0; kb < nk; ++kb) {
    BV32 av, bv;
    av.q[0] = *(const uint4*)(arow + (kb << 5));
    av.q[1] = *(const uint4*)(arow + (kb << 5) + 16);
    const uint4* bp = (const uint4*)(bq + ((size_t)kb << 9));
    bv.q[0] = bp[0];
    bv.q[1] = bp[1];
    acc = __builtin_amdgcn_wmma_f32_16x16x32_bf16(
        false, av.v, false, bv.v, (short)0, acc, false, false);
  }
  const int n = (nT << 4) + lm;
  const int mBase = (mT << 4) + (half << 3);
#pragma unroll
  for (int v = 0; v < 8; ++v)
    C[(size_t)(mBase + v) * (size_t)Nout + n] = acc[v];
}

// ---------------- edge scatter: AGG[dst] += H[src] * dis[src]*dis[dst] ----------------
__global__ void k_scatter(const int* __restrict__ src, const int* __restrict__ dst,
                          const float* __restrict__ dis, const float* __restrict__ H,
                          float* __restrict__ AGG, int E, int Dout) {
  long t = (long)blockIdx.x * blockDim.x + threadIdx.x;
  int chunks = Dout >> 2;
  long total = (long)E * chunks;
  if (t >= total) return;
  int e = (int)(t / chunks);
  int c = (int)(t - (long)e * chunks) << 2;
  int s = src[e], d = dst[e];
  float w = dis[s] * dis[d];
  const float4 hv = *(const float4*)(H + (size_t)s * Dout + c);
  float* ap = AGG + (size_t)d * Dout + c;
  atomicAdd(ap + 0, hv.x * w);
  atomicAdd(ap + 1, hv.y * w);
  atomicAdd(ap + 2, hv.z * w);
  atomicAdd(ap + 3, hv.w * w);
}

// act = relu(AGG + H*dis^2 + b); write back f32 (in AGG) and bf16 (next layer A)
__global__ void k_finalize(float* __restrict__ AGG, const float* __restrict__ H,
                           const float* __restrict__ dis, const float* __restrict__ b,
                           unsigned short* __restrict__ BFout, int N, int Dout) {
  long idx = (long)blockIdx.x * blockDim.x + threadIdx.x;
  long total = (long)N * Dout;
  if (idx >= total) return;
  int i = (int)(idx / Dout);
  int j = (int)(idx - (long)i * Dout);
  float self = dis[i] * dis[i];
  float v = AGG[idx] + H[idx] * self + b[j];
  v = fmaxf(v, 0.0f);
  AGG[idx] = v;
  BFout[idx] = f2bf(v);
}

// ---------------- pooling ----------------
__global__ void k_count(const int* __restrict__ batch, float* __restrict__ cnt, int n) {
  int i = blockIdx.x * blockDim.x + threadIdx.x;
  if (i < n) atomicAdd(&cnt[batch[i]], 1.0f);
}

__global__ void k_pool(const float* __restrict__ act, const int* __restrict__ batch,
                       float* __restrict__ s, int N, int F) {
  long idx = (long)blockIdx.x * blockDim.x + threadIdx.x;
  long total = (long)N * F;
  if (idx >= total) return;
  int i = (int)(idx / F);
  int j = (int)(idx - (long)i * F);
  atomicAdd(&s[(size_t)batch[i] * F + j], act[idx]);
}

__global__ void k_mean(float* __restrict__ s, const float* __restrict__ cnt, int G, int F) {
  int idx = blockIdx.x * blockDim.x + threadIdx.x;
  if (idx >= G * F) return;
  int g = idx / F;
  s[idx] = s[idx] / fmaxf(cnt[g], 1.0f);
}

// ---------------- small fp32 linear: out[R,Nout] = (relu?)(in[R,K] @ W[K,Nout] + b) ----------------
__global__ void k_linear(const float* __restrict__ in, const float* __restrict__ W,
                         const float* __restrict__ b, float* __restrict__ out,
                         int R, int K, int Nout, int relu) {
  int idx = blockIdx.x * blockDim.x + threadIdx.x;
  if (idx >= R * Nout) return;
  int r = idx / Nout;
  int n = idx - r * Nout;
  float acc = b[n];
  const float* ip = in + (size_t)r * K;
  for (int k = 0; k < K; ++k) acc += ip[k] * W[(size_t)k * Nout + n];
  if (relu) acc = fmaxf(acc, 0.0f);
  out[idx] = acc;
}

__global__ void k_concat(const float* __restrict__ fp, const float* __restrict__ xt,
                         float* __restrict__ xc, int G, int FP, int FT) {
  int idx = blockIdx.x * blockDim.x + threadIdx.x;
  int F = FP + FT;
  if (idx >= G * F) return;
  int g = idx / F;
  int j = idx - g * F;
  xc[idx] = (j < FP) ? fp[(size_t)g * FP + j] : xt[(size_t)g * FT + (j - FP)];
}

// ---------------- host-side orchestration ----------------
extern "C" void kernel_launch(void* const* d_in, const int* in_sizes, int n_in,
                              void* d_out, int out_size, void* d_ws, size_t ws_size,
                              hipStream_t stream) {
  (void)n_in; (void)out_size; (void)ws_size;
  const float* x     = (const float*)d_in[0];
  const int*   edge  = (const int*)d_in[1];
  const int*   batch = (const int*)d_in[2];
  const float* fp_x  = (const float*)d_in[3];
  const float* W1 = (const float*)d_in[4];  const float* b1 = (const float*)d_in[5];
  const float* W2 = (const float*)d_in[6];  const float* b2 = (const float*)d_in[7];
  const float* W3 = (const float*)d_in[8];  const float* b3 = (const float*)d_in[9];
  const float* Wg1 = (const float*)d_in[10]; const float* bg1 = (const float*)d_in[11];
  const float* Wg2 = (const float*)d_in[12]; const float* bg2 = (const float*)d_in[13];
  const float* Wf1 = (const float*)d_in[14]; const float* bf1 = (const float*)d_in[15];
  const float* Wf2 = (const float*)d_in[16]; const float* bf2 = (const float*)d_in[17];
  const float* Wo  = (const float*)d_in[18]; const float* bo  = (const float*)d_in[19];

  const int D1 = 1280, D2 = 640, D3 = 320, FPDIM = 65;
  const int N = in_sizes[0] / D1;
  const int E = in_sizes[1] / 2;
  const int G = in_sizes[3] / FPDIM;
  const int* esrc = edge;
  const int* edst = edge + E;

  // workspace carve-out (256B aligned)
  char* ws = (char*)d_ws;
  size_t off = 0;
  auto alloc = [&](size_t bytes) {
    size_t o = off;
    off += (bytes + 255) & ~(size_t)255;
    return o;
  };
  float*          dis  = (float*)(ws + alloc((size_t)N * 4));
  unsigned short* WP   = (unsigned short*)(ws + alloc((size_t)D1 * D1 * 2));
  unsigned short* BFA  = (unsigned short*)(ws + alloc((size_t)N * D1 * 2));
  float*          H    = (float*)(ws + alloc((size_t)N * D1 * 4));
  float*          AGG  = (float*)(ws + alloc((size_t)N * D1 * 4));
  float*          pool = (float*)(ws + alloc((size_t)G * D3 * 4));
  float*          cnt  = (float*)(ws + alloc((size_t)G * 4));
  float*          xt1  = (float*)(ws + alloc((size_t)G * 1024 * 4));
  float*          xt2  = (float*)(ws + alloc((size_t)G * 128 * 4));
  float*          xc   = (float*)(ws + alloc((size_t)G * 193 * 4));
  float*          f1   = (float*)(ws + alloc((size_t)G * 1024 * 4));
  float*          f2   = (float*)(ws + alloc((size_t)G * 512 * 4));

  const int BT = 256;

  // degree -> dis = 1/sqrt(deg), deg includes self loop
  k_fill<<<CDIV(N, BT), BT, 0, stream>>>(dis, N, 1.0f);
  k_deg<<<CDIV(E, BT), BT, 0, stream>>>(edst, dis, E);
  k_invsqrt<<<CDIV(N, BT), BT, 0, stream>>>(dis, N);

  // x -> bf16
  {
    long n = (long)N * D1;
    k_f2bf<<<CDIV(n, (long)BT), BT, 0, stream>>>(x, BFA, n);
  }

  auto gcn_layer = [&](const float* W, const float* b, int Din, int Dout) {
    long wtot = (long)Din * Dout;
    k_packW<<<CDIV(wtot, (long)BT), BT, 0, stream>>>(W, WP, Din, Dout);
    const int tilesN = Dout / 16;
    if ((tilesN & 1) == 0) {
      const int groupsM = N / (16 * MSUB);
      int waves = groupsM * (tilesN / NSUB);
      k_gemm_bf16<<<CDIV(waves * 32, BT), BT, 0, stream>>>(BFA, WP, H, N, Din, Dout);
      int remRows = N - groupsM * (16 * MSUB);
      if (remRows > 0) {
        int rwaves = (remRows / 16) * tilesN;
        k_gemm_bf16_1<<<CDIV(rwaves * 32, BT), BT, 0, stream>>>(
            BFA + (size_t)groupsM * (16 * MSUB) * Din, WP,
            H + (size_t)groupsM * (16 * MSUB) * Dout, remRows, Din, Dout);
      }
    } else {
      int waves = (N / 16) * tilesN;
      k_gemm_bf16_1<<<CDIV(waves * 32, BT), BT, 0, stream>>>(BFA, WP, H, N, Din, Dout);
    }
    long ne = (long)N * Dout;
    k_fill<<<CDIV(ne, (long)BT), BT, 0, stream>>>(AGG, ne, 0.0f);
    long st = (long)E * (Dout >> 2);
    k_scatter<<<CDIV(st, (long)BT), BT, 0, stream>>>(esrc, edst, dis, H, AGG, E, Dout);
    k_finalize<<<CDIV(ne, (long)BT), BT, 0, stream>>>(AGG, H, dis, b, BFA, N, Dout);
  };

  gcn_layer(W1, b1, D1, D1);
  gcn_layer(W2, b2, D1, D2);
  gcn_layer(W3, b3, D2, D3);

  // global mean pool (act3 lives in AGG, fp32)
  k_fill<<<CDIV((long)G * D3, (long)BT), BT, 0, stream>>>(pool, (long)G * D3, 0.0f);
  k_fill<<<CDIV(G, BT), BT, 0, stream>>>(cnt, G, 0.0f);
  k_count<<<CDIV(N, BT), BT, 0, stream>>>(batch, cnt, N);
  k_pool<<<CDIV((long)N * D3, (long)BT), BT, 0, stream>>>(AGG, batch, pool, N, D3);
  k_mean<<<CDIV(G * D3, BT), BT, 0, stream>>>(pool, cnt, G, D3);

  // MLP head (fp32)
  k_linear<<<CDIV(G * 1024, BT), BT, 0, stream>>>(pool, Wg1, bg1, xt1, G, D3, 1024, 1);
  k_linear<<<CDIV(G * 128, BT), BT, 0, stream>>>(xt1, Wg2, bg2, xt2, G, 1024, 128, 0);
  k_concat<<<CDIV(G * 193, BT), BT, 0, stream>>>(fp_x, xt2, xc, G, FPDIM, 128);
  k_linear<<<CDIV(G * 1024, BT), BT, 0, stream>>>(xc, Wf1, bf1, f1, G, 193, 1024, 1);
  k_linear<<<CDIV(G * 512, BT), BT, 0, stream>>>(f1, Wf2, bf2, f2, G, 1024, 512, 1);
  k_linear<<<CDIV(G * 1, BT), BT, 0, stream>>>(f2, Wo, bo, (float*)d_out, G, 512, 1, 0);
}
